// GNN_22445499088918
// MI455X (gfx1250) — matrix operand order repaired and checked by reading, not compiled, not run on hardware
//
#include <hip/hip_runtime.h>
#include <hip/hip_bf16.h>

#define N_NODES 50000
#define N_EDGES 1600000
#define N_GRAPHS 2048
#define HID 64
#define BN_EPS 1e-5f

typedef __attribute__((ext_vector_type(16))) _Float16 v16h;
typedef __attribute__((ext_vector_type(8)))  float    v8f;

// ---------------------------------------------------------------------------
// K0: fuse edge_encoder + edge_lin:  Wfe[4][64], bfe[64]
// ---------------------------------------------------------------------------
__global__ void fuse_edge_weights(const float* __restrict__ Wew,   // [4,12]
                                  const float* __restrict__ Web,   // [12]
                                  const float* __restrict__ Welw,  // [12,64]
                                  const float* __restrict__ Welb,  // [64]
                                  float* __restrict__ Wfe,         // [4,64]
                                  float* __restrict__ bfe) {       // [64]
  int t = threadIdx.x;
  int k = t >> 6, j = t & 63;
  float acc = 0.f;
  #pragma unroll
  for (int i = 0; i < 12; ++i) acc += Wew[k * 12 + i] * Welw[i * 64 + j];
  Wfe[k * 64 + j] = acc;
  if (t < 64) {
    float b = Welb[t];
    #pragma unroll
    for (int i = 0; i < 12; ++i) b += Web[i] * Welw[i * 64 + t];
    bfe[t] = b;
  }
}

// ---------------------------------------------------------------------------
// zero fill
// ---------------------------------------------------------------------------
__global__ void zero_kernel(float* __restrict__ p, int n) {
  int i = blockIdx.x * blockDim.x + threadIdx.x;
  int stride = gridDim.x * blockDim.x;
  for (; i < n; i += stride) p[i] = 0.f;
}

// ---------------------------------------------------------------------------
// K2: conv1 messages (feature dim 12). One thread per edge.
// ---------------------------------------------------------------------------
__global__ void conv1_msg(const float* __restrict__ x,        // [N,12]
                          const int*   __restrict__ ei,       // [2,E]
                          const float* __restrict__ eattr,    // [E,4]
                          const float* __restrict__ Wew,      // [4,12]
                          const float* __restrict__ Web,      // [12]
                          float* __restrict__ agg) {          // [N,12]
  __shared__ float Ws[48];
  __shared__ float Bs[12];
  if (threadIdx.x < 48) Ws[threadIdx.x] = Wew[threadIdx.x];
  if (threadIdx.x < 12) Bs[threadIdx.x] = Web[threadIdx.x];
  __syncthreads();
  int e = blockIdx.x * blockDim.x + threadIdx.x;
  if (e >= N_EDGES) return;
  float4 ea = ((const float4*)eattr)[e];
  int s = ei[e];
  int d = ei[N_EDGES + e];
  #pragma unroll
  for (int i = 0; i < 12; ++i) {
    float e1 = Bs[i] + ea.x * Ws[i] + ea.y * Ws[12 + i] + ea.z * Ws[24 + i] + ea.w * Ws[36 + i];
    float m = fmaxf(x[s * 12 + i] + e1, 0.f);
    unsafeAtomicAdd(&agg[d * 12 + i], m);
  }
}

// ---------------------------------------------------------------------------
// K7: conv2/3 messages (feature dim 64). 8 threads per edge, 8 cols each.
// e_deep recomputed on the fly from fused [4,64] weight (saves 410MB x2 HBM).
// ---------------------------------------------------------------------------
__global__ void conv_msg64(const float* __restrict__ h,       // [N,64]
                           const int*   __restrict__ ei,      // [2,E]
                           const float* __restrict__ eattr,   // [E,4]
                           const float* __restrict__ Wfe,     // [4,64]
                           const float* __restrict__ bfe,     // [64]
                           float* __restrict__ agg) {         // [N,64]
  __shared__ float Ws[256];
  __shared__ float Bs[64];
  if (threadIdx.x < 256) Ws[threadIdx.x] = Wfe[threadIdx.x];
  if (threadIdx.x < 64)  Bs[threadIdx.x] = bfe[threadIdx.x];
  __syncthreads();
  long gid = (long)blockIdx.x * blockDim.x + threadIdx.x;
  long e = gid >> 3;
  int  g = (int)(gid & 7);
  if (e >= N_EDGES) return;
  float4 ea = ((const float4*)eattr)[e];
  int s = ei[e];
  int d = ei[N_EDGES + e];
  int j0 = g * 8;
  #pragma unroll
  for (int i = 0; i < 8; ++i) {
    int j = j0 + i;
    float ed = Bs[j] + ea.x * Ws[j] + ea.y * Ws[64 + j] + ea.z * Ws[128 + j] + ea.w * Ws[192 + j];
    float m = fmaxf(h[(long)s * 64 + j] + ed, 0.f);
    unsafeAtomicAdd(&agg[(long)d * 64 + j], m);
  }
}

// ---------------------------------------------------------------------------
// Fused node MLP: out = relu((x+agg) @ W1 + b1) @ W2 + b2
// K1 = input dim (12 or 64). WMMA f32_16x16x32_f16, f32 accumulate.
// Block: 256 threads = 8 waves, 128 rows. Each wave: 16 rows x 64 cols.
// ---------------------------------------------------------------------------
#define A_ST 72   // LDS row stride in f16 (144B, 16B aligned)

__global__ void node_mlp(const float* __restrict__ xin,  // [N,K1]
                         const float* __restrict__ agg,  // [N,K1]
                         int K1,
                         const float* __restrict__ W1,   // [K1,64]
                         const float* __restrict__ b1,   // [64]
                         const float* __restrict__ W2,   // [64,64]
                         const float* __restrict__ b2,   // [64]
                         float* __restrict__ out,        // [N,64]
                         int n) {
  __shared__ __align__(16) _Float16 Asm[128 * A_ST];
  __shared__ __align__(16) _Float16 Hsm[128 * A_ST];
  __shared__ __align__(16) _Float16 W1s[64 * 64];  // transposed: [n][k]
  __shared__ __align__(16) _Float16 W2s[64 * 64];  // transposed: [n][k]
  __shared__ float b1s[64], b2s[64];

  int tid = threadIdx.x;
  // stage weights (transposed so B fragments are contiguous in k)
  for (int idx = tid; idx < 64 * 64; idx += 256) {
    int k = idx >> 6, nn = idx & 63;
    W1s[nn * 64 + k] = (k < K1) ? (_Float16)W1[k * 64 + nn] : (_Float16)0.f;
    W2s[nn * 64 + k] = (_Float16)W2[k * 64 + nn];
  }
  if (tid < 64) { b1s[tid] = b1[tid]; b2s[tid] = b2[tid]; }
  // stage A = x + agg, converted to f16, zero-padded to 64 cols
  int r0 = blockIdx.x * 128;
  for (int idx = tid; idx < 128 * 64; idx += 256) {
    int r = idx >> 6, k = idx & 63;
    int row = r0 + r;
    float v = 0.f;
    if (row < n && k < K1) v = xin[(long)row * K1 + k] + agg[(long)row * K1 + k];
    Asm[r * A_ST + k] = (_Float16)v;
  }
  __syncthreads();

  int lane = tid & 31;
  int wave = tid >> 5;
  int wr   = wave * 16;          // wave's row block within the 128
  int m    = lane & 15;
  int half = lane >> 4;

  int ksteps1 = (K1 > 32) ? 2 : 1;

  // ---- GEMM 1: hidden = relu(A @ W1 + b1) ----
  v8f acc[4];
  #pragma unroll
  for (int t = 0; t < 4; ++t) acc[t] = (v8f)(0.f);
  for (int ks = 0; ks < ksteps1; ++ks) {
    int kk = ks * 32;
    v16h af;
    int abase = (wr + m) * A_ST + kk + 16 * half;
    #pragma unroll
    for (int i = 0; i < 8; ++i) { af[i] = Asm[abase + i]; af[8 + i] = Asm[abase + 8 + i]; }
    #pragma unroll
    for (int t = 0; t < 4; ++t) {
      int n0 = t * 16 + m;
      int kb = kk + 16 * half;
      v16h bf;
      #pragma unroll
      for (int i = 0; i < 16; ++i) bf[i] = W1s[n0 * 64 + kb + i];
      acc[t] = __builtin_amdgcn_wmma_f32_16x16x32_f16(false, af, false, bf,
                                                      (short)0, acc[t], false, false);
    }
  }
  // write hidden (relu + bias) into LDS as f16
  #pragma unroll
  for (int t = 0; t < 4; ++t) {
    #pragma unroll
    for (int v = 0; v < 8; ++v) {
      int mm = v + 8 * half;
      int nn = t * 16 + m;
      float hv = fmaxf(acc[t][v] + b1s[nn], 0.f);
      Hsm[(wr + mm) * A_ST + nn] = (_Float16)hv;
    }
  }
  __syncthreads();

  // ---- GEMM 2: out = hidden @ W2 + b2 ----
  v8f acc2[4];
  #pragma unroll
  for (int t = 0; t < 4; ++t) acc2[t] = (v8f)(0.f);
  #pragma unroll
  for (int ks = 0; ks < 2; ++ks) {
    int kk = ks * 32;
    v16h af;
    int abase = (wr + m) * A_ST + kk + 16 * half;
    #pragma unroll
    for (int i = 0; i < 8; ++i) { af[i] = Hsm[abase + i]; af[8 + i] = Hsm[abase + 8 + i]; }
    #pragma unroll
    for (int t = 0; t < 4; ++t) {
      int n0 = t * 16 + m;
      int kb = kk + 16 * half;
      v16h bf;
      #pragma unroll
      for (int i = 0; i < 16; ++i) bf[i] = W2s[n0 * 64 + kb + i];
      acc2[t] = __builtin_amdgcn_wmma_f32_16x16x32_f16(false, af, false, bf,
                                                       (short)0, acc2[t], false, false);
    }
  }
  // store fp32
  #pragma unroll
  for (int t = 0; t < 4; ++t) {
    #pragma unroll
    for (int v = 0; v < 8; ++v) {
      int mm = v + 8 * half;
      int nn = t * 16 + m;
      int row = r0 + wr + mm;
      if (row < n) out[(long)row * 64 + nn] = acc2[t][v] + b2s[nn];
    }
  }
}

// ---------------------------------------------------------------------------
// BN stats: per-column sum & sumsq over N rows
// ---------------------------------------------------------------------------
__global__ void bn_stats(const float* __restrict__ h, float* __restrict__ sums,
                         float* __restrict__ sq, int n) {
  __shared__ float ls[256], lq[256];
  int c = threadIdx.x & 63;
  int rsub = threadIdx.x >> 6;   // 0..3
  float s = 0.f, q = 0.f;
  for (int row = blockIdx.x * 4 + rsub; row < n; row += gridDim.x * 4) {
    float v = h[(long)row * 64 + c];
    s += v; q += v * v;
  }
  ls[threadIdx.x] = s; lq[threadIdx.x] = q;
  __syncthreads();
  if (threadIdx.x < 64) {
    float ts = ls[threadIdx.x] + ls[threadIdx.x + 64] + ls[threadIdx.x + 128] + ls[threadIdx.x + 192];
    float tq = lq[threadIdx.x] + lq[threadIdx.x + 64] + lq[threadIdx.x + 128] + lq[threadIdx.x + 192];
    unsafeAtomicAdd(&sums[threadIdx.x], ts);
    unsafeAtomicAdd(&sq[threadIdx.x], tq);
  }
}

// ---------------------------------------------------------------------------
// BN apply + relu, in place
// ---------------------------------------------------------------------------
__global__ void bn_apply(float* __restrict__ h, const float* __restrict__ sums,
                         const float* __restrict__ sq, const float* __restrict__ gamma,
                         const float* __restrict__ beta, int n) {
  long idx = (long)blockIdx.x * blockDim.x + threadIdx.x;
  if (idx >= (long)n * 64) return;
  int c = (int)(idx & 63);
  float inv_n = 1.f / (float)n;
  float mu  = sums[c] * inv_n;
  float var = sq[c] * inv_n - mu * mu;
  float sc  = rsqrtf(var + BN_EPS) * gamma[c];
  float v   = (h[idx] - mu) * sc + beta[c];
  h[idx] = fmaxf(v, 0.f);
}

// ---------------------------------------------------------------------------
// global mean pool: accumulate sums + counts
// ---------------------------------------------------------------------------
__global__ void pool_kernel(const float* __restrict__ h, const int* __restrict__ batch,
                            float* __restrict__ pooled, float* __restrict__ cnts, int n) {
  long idx = (long)blockIdx.x * blockDim.x + threadIdx.x;
  if (idx >= (long)n * 64) return;
  int row = (int)(idx >> 6);
  int c   = (int)(idx & 63);
  int b = batch[row];
  unsafeAtomicAdd(&pooled[(long)b * 64 + c], h[idx]);
  if (c == 0) unsafeAtomicAdd(&cnts[b], 1.f);
}

// ---------------------------------------------------------------------------
// head: out[g] = relu(pooled/cnt @ Wh1 + bh1) @ Wh2 + bh2
// ---------------------------------------------------------------------------
__global__ void head_kernel(const float* __restrict__ pooled, const float* __restrict__ cnts,
                            const float* __restrict__ Wh1, const float* __restrict__ bh1,
                            const float* __restrict__ Wh2, const float* __restrict__ bh2,
                            float* __restrict__ outp) {
  __shared__ float W1s[64 * 32];
  __shared__ float b1s[32], W2s[32];
  for (int idx = threadIdx.x; idx < 64 * 32; idx += 256) W1s[idx] = Wh1[idx];
  if (threadIdx.x < 32) { b1s[threadIdx.x] = bh1[threadIdx.x]; W2s[threadIdx.x] = Wh2[threadIdx.x]; }
  __syncthreads();
  int g = blockIdx.x * blockDim.x + threadIdx.x;
  if (g >= N_GRAPHS) return;
  float cnt = fmaxf(cnts[g], 1.f);
  float inv = 1.f / cnt;
  float p[64];
  #pragma unroll
  for (int i = 0; i < 64; ++i) p[i] = pooled[(long)g * 64 + i] * inv;
  float acc = bh2[0];
  for (int j = 0; j < 32; ++j) {
    float hj = b1s[j];
    #pragma unroll
    for (int i = 0; i < 64; ++i) hj += p[i] * W1s[i * 32 + j];
    acc += fmaxf(hj, 0.f) * W2s[j];
  }
  outp[g] = acc;
}

// ---------------------------------------------------------------------------
static inline void launch_zero(float* p, long n, hipStream_t s) {
  int blocks = (int)((n + 255) / 256);
  if (blocks > 4096) blocks = 4096;
  zero_kernel<<<blocks, 256, 0, s>>>(p, (int)n);
}

extern "C" void kernel_launch(void* const* d_in, const int* in_sizes, int n_in,
                              void* d_out, int out_size, void* d_ws, size_t ws_size,
                              hipStream_t stream) {
  const float* x     = (const float*)d_in[0];
  const int*   ei    = (const int*)d_in[1];
  const float* eattr = (const float*)d_in[2];
  const int*   batch = (const int*)d_in[3];
  const float* Wew   = (const float*)d_in[4];
  const float* Web   = (const float*)d_in[5];
  const float* W1a   = (const float*)d_in[6];
  const float* b1a   = (const float*)d_in[7];
  const float* W1b   = (const float*)d_in[8];
  const float* b1b   = (const float*)d_in[9];
  const float* g1    = (const float*)d_in[10];
  const float* be1   = (const float*)d_in[11];
  const float* Welw  = (const float*)d_in[12];
  const float* Welb  = (const float*)d_in[13];
  const float* W2a   = (const float*)d_in[14];
  const float* b2a   = (const float*)d_in[15];
  const float* W2b   = (const float*)d_in[16];
  const float* b2b   = (const float*)d_in[17];
  const float* g2    = (const float*)d_in[18];
  const float* be2   = (const float*)d_in[19];
  const float* W3a   = (const float*)d_in[20];
  const float* b3a   = (const float*)d_in[21];
  const float* W3b   = (const float*)d_in[22];
  const float* b3b   = (const float*)d_in[23];
  const float* g3    = (const float*)d_in[24];
  const float* be3   = (const float*)d_in[25];
  const float* Wh1   = (const float*)d_in[26];
  const float* bh1   = (const float*)d_in[27];
  const float* Wh2   = (const float*)d_in[28];
  const float* bh2   = (const float*)d_in[29];
  float* outp = (float*)d_out;

  // workspace layout
  const long N64 = (long)N_NODES * 64;
  float* ws    = (float*)d_ws;
  float* h_a   = ws;                 // [N,64]
  float* h_b   = h_a + N64;          // [N,64]
  float* agg   = h_b + N64;          // [N,64]
  float* pooled= agg + N64;          // [G,64]
  float* cnts  = pooled + (long)N_GRAPHS * 64;  // [G]
  float* bnsum = cnts + N_GRAPHS;    // [64]
  float* bnsq  = bnsum + 64;         // [64]
  float* Wfe   = bnsq + 64;          // [4,64]
  float* bfe   = Wfe + 256;          // [64]

  const int mlp_blocks = (N_NODES + 127) / 128;
  const int e1_blocks  = (N_EDGES + 255) / 256;
  const int e8_blocks  = (int)(((long)N_EDGES * 8 + 255) / 256);
  const int nc_blocks  = (int)((N64 + 255) / 256);

  // fused edge weights
  fuse_edge_weights<<<1, 256, 0, stream>>>(Wew, Web, Welw, Welb, Wfe, bfe);

  // ---- conv1 ----
  launch_zero(agg, (long)N_NODES * 12, stream);
  conv1_msg<<<e1_blocks, 256, 0, stream>>>(x, ei, eattr, Wew, Web, agg);
  node_mlp<<<mlp_blocks, 256, 0, stream>>>(x, agg, 12, W1a, b1a, W1b, b1b, h_a, N_NODES);
  launch_zero(bnsum, 128, stream);
  bn_stats<<<128, 256, 0, stream>>>(h_a, bnsum, bnsq, N_NODES);
  bn_apply<<<nc_blocks, 256, 0, stream>>>(h_a, bnsum, bnsq, g1, be1, N_NODES);

  // ---- conv2 ----
  launch_zero(agg, N64, stream);
  conv_msg64<<<e8_blocks, 256, 0, stream>>>(h_a, ei, eattr, Wfe, bfe, agg);
  node_mlp<<<mlp_blocks, 256, 0, stream>>>(h_a, agg, 64, W2a, b2a, W2b, b2b, h_b, N_NODES);
  launch_zero(bnsum, 128, stream);
  bn_stats<<<128, 256, 0, stream>>>(h_b, bnsum, bnsq, N_NODES);
  bn_apply<<<nc_blocks, 256, 0, stream>>>(h_b, bnsum, bnsq, g2, be2, N_NODES);

  // ---- conv3 ----
  launch_zero(agg, N64, stream);
  conv_msg64<<<e8_blocks, 256, 0, stream>>>(h_b, ei, eattr, Wfe, bfe, agg);
  node_mlp<<<mlp_blocks, 256, 0, stream>>>(h_b, agg, 64, W3a, b3a, W3b, b3b, h_a, N_NODES);
  launch_zero(bnsum, 128, stream);
  bn_stats<<<128, 256, 0, stream>>>(h_a, bnsum, bnsq, N_NODES);
  bn_apply<<<nc_blocks, 256, 0, stream>>>(h_a, bnsum, bnsq, g3, be3, N_NODES);

  // ---- pool + head ----
  launch_zero(pooled, (long)N_GRAPHS * 64 + N_GRAPHS, stream);
  pool_kernel<<<nc_blocks, 256, 0, stream>>>(h_a, batch, pooled, cnts, N_NODES);
  head_kernel<<<N_GRAPHS / 256, 256, 0, stream>>>(pooled, cnts, Wh1, bh1, Wh2, bh2, outp);
}
